// CausalSelfAttention_26139170964311
// MI455X (gfx1250) — compile-verified, matrix operand order checked by reading, not run on hardware
//
#include <hip/hip_runtime.h>
#include <hip/hip_bf16.h>

// Problem constants (match reference)
static constexpr int Bn   = 2;
static constexpr int T_   = 2048;
static constexpr int C_   = 1024;
static constexpr int HQ_  = 16;
static constexpr int HKV_ = 2;
static constexpr int HD_  = 64;

typedef __attribute__((ext_vector_type(16))) __bf16 v16bf;
typedef __attribute__((ext_vector_type(8)))  float  v8f;

// ---------------------------------------------------------------------------
// f32 -> bf16 bulk conversion (one-time pass; n must be a multiple of 8)
// ---------------------------------------------------------------------------
__global__ void __launch_bounds__(256)
cvt_f32_bf16_kernel(const float* __restrict__ src, __bf16* __restrict__ dst, int n) {
  const int i = (blockIdx.x * 256 + threadIdx.x) * 8;
  if (i < n) {
#pragma unroll
    for (int j = 0; j < 8; ++j) dst[i + j] = (__bf16)src[i + j];
  }
}

// ---------------------------------------------------------------------------
// 16x32 bf16 fragment loader (CDNA5 ISA 7.12.2, 16-bit A 16x32 layout).
// Per-lane data is two contiguous 16-byte chunks:
//   elements 0..7  <-> K =      8*half + 0..7
//   elements 8..15 <-> K = 16 + 8*half + 0..7
// For the B operand pass the "B-transposed" (N x K, row-major) matrix; the
// same layout applies with M->N.
// ---------------------------------------------------------------------------
__device__ __forceinline__ v16bf load_frag16x32(const __bf16* __restrict__ src, int ld) {
  const int lane = threadIdx.x & 31;
  const int half = lane >> 4;
  const int m    = lane & 15;
  const __bf16* p = src + (size_t)m * ld + 8 * half;
  v16bf f;
#pragma unroll
  for (int i = 0; i < 8; ++i) {
    f[i]     = p[i];       // merges into global_load_b128
    f[8 + i] = p[16 + i];  // merges into global_load_b128
  }
  return f;
}

__device__ __forceinline__ v8f wmma_bf16(v16bf a, v16bf b, v8f c) {
  // (neg_a, A, neg_b, B, c_mod, C, reuse_a, reuse_b)
  return __builtin_amdgcn_wmma_f32_16x16x32_bf16(false, a, false, b, (short)0, c,
                                                 false, false);
}

// ---------------------------------------------------------------------------
// Kernel 1: fused QKV projection + bias + RoPE (+ q pre-scale 1/HD).
// 32x64 output tile per wave: 2 A-frags x 4 B-frags per K-step -> 8 WMMAs,
// each B-frag reused by both row-halves.  64-wide tile == one head.
//   q : bf16 [B][HQ ][T][HD]   (rope'd, scaled by 1/64 = scale^2)
//   k : bf16 [B][HKV][T][HD]   (rope'd)
//   vT: bf16 [B][HKV][HD][T]   (transposed so P@V B-frags are contiguous)
// grid = (B*T/32, 16(Q)+2(K)+2(V)), block = 32 (1 wave)
// ---------------------------------------------------------------------------
__global__ void __launch_bounds__(32)
qkv_rope_kernel(const __bf16* __restrict__ xb, const float* __restrict__ rope,
                const __bf16* __restrict__ Wqb, const float* __restrict__ bq,
                const __bf16* __restrict__ Wkb, const float* __restrict__ bk,
                const __bf16* __restrict__ Wvb, const float* __restrict__ bv,
                __bf16* __restrict__ q_out, __bf16* __restrict__ k_out,
                __bf16* __restrict__ vT_out) {
  const int lane = threadIdx.x & 31;
  const int half = lane >> 4;
  const int col  = lane & 15;
  const int rt   = blockIdx.x;  // 32-token row tile over B*T
  const int ct   = blockIdx.y;  // 64-wide output feature tile

  int kind, n0;
  const __bf16* W;
  const float*  bias;
  if (ct < 16)      { kind = 0; n0 = ct * 64;        W = Wqb; bias = bq; }
  else if (ct < 18) { kind = 1; n0 = (ct - 16) * 64; W = Wkb; bias = bk; }
  else              { kind = 2; n0 = (ct - 18) * 64; W = Wvb; bias = bv; }

  const __bf16* xa = xb + (size_t)rt * 32 * C_;
  const __bf16* wb = W + (size_t)n0 * C_;   // row n of W == column n of W^T

  v8f acc[2][4] = {{{}, {}, {}, {}}, {{}, {}, {}, {}}};
#pragma unroll 2
  for (int kk = 0; kk < C_; kk += 32) {
    const v16bf a0 = load_frag16x32(xa + kk, C_);
    const v16bf a1 = load_frag16x32(xa + (size_t)16 * C_ + kk, C_);
#pragma unroll
    for (int s = 0; s < 4; ++s) {
      const v16bf w = load_frag16x32(wb + (size_t)s * 16 * C_ + kk, C_);
      acc[0][s] = wmma_bf16(a0, w, acc[0][s]);
      acc[1][s] = wmma_bf16(a1, w, acc[1][s]);
    }
  }

  const int head = n0 >> 6;  // tile is head-aligned (64-wide)
#pragma unroll
  for (int mh = 0; mh < 2; ++mh) {
#pragma unroll
    for (int s = 0; s < 4; ++s) {
      const int   d  = s * 16 + col;     // feature within head
      const float bb = bias[n0 + d];
#pragma unroll
      for (int j = 0; j < 8; ++j) {
        const int row = rt * 32 + mh * 16 + j + 8 * half;  // token over B*T
        const int b_  = row / T_;
        const int t   = row - b_ * T_;
        float val = acc[mh][s][j] + bb;
        if (kind == 2) {
          vT_out[(((size_t)b_ * HKV_ + head) * HD_ + d) * T_ + t] = (__bf16)val;
        } else {
          // RoPE: the (even,odd) feature pair sits in adjacent lanes
          const float other = __shfl_xor(val, 1, 32);
          const float sn = rope[t * HD_ + (d & ~1)];
          const float cs = rope[t * HD_ + (d | 1)];
          float o = (d & 1) ? (val * cs + other * sn)   // o1 = x1*cos + x0*sin
                            : (val * cs - other * sn);  // o0 = x0*cos - x1*sin
          if (kind == 0) {
            o *= 0.015625f;  // scale^2 = 1/HD (q*scale and scores*scale folded)
            q_out[(((size_t)b_ * HQ_ + head) * T_ + t) * HD_ + d] = (__bf16)o;
          } else {
            k_out[(((size_t)b_ * HKV_ + head) * T_ + t) * HD_ + d] = (__bf16)o;
          }
        }
      }
    }
  }
}

// ---------------------------------------------------------------------------
// Kernel 2: flash attention (causal, online softmax), 32 q-rows per wave,
// 32 keys per block-iteration; k/v B-frags reused by both q row-halves.
// grid = (B*HQ, T/32), block = 32.
// ---------------------------------------------------------------------------
__global__ void __launch_bounds__(32)
attn_kernel(const __bf16* __restrict__ qg, const __bf16* __restrict__ kg,
            const __bf16* __restrict__ vTg, __bf16* __restrict__ yg) {
  __shared__ __bf16 lds[2 * 16 * 32];  // P tiles (D-layout -> A-layout), 2 KB

  const int lane = threadIdx.x & 31;
  const int half = lane >> 4;
  const int col  = lane & 15;
  const int bh   = blockIdx.x;
  const int b    = bh / HQ_;
  const int h    = bh - b * HQ_;
  const int hkv  = h / (HQ_ / HKV_);
  const int m0   = blockIdx.y * 32;

  const __bf16* qtile = qg + (((size_t)b * HQ_ + h) * T_ + m0) * HD_;
  v16bf qa[2][2];
#pragma unroll
  for (int mh = 0; mh < 2; ++mh) {
    qa[mh][0] = load_frag16x32(qtile + (size_t)mh * 16 * HD_, HD_);       // K 0..31
    qa[mh][1] = load_frag16x32(qtile + (size_t)mh * 16 * HD_ + 32, HD_);  // K 32..63
  }

  const __bf16* kbase = kg + ((size_t)b * HKV_ + hkv) * T_ * HD_;
  const __bf16* vbase = vTg + ((size_t)b * HKV_ + hkv) * HD_ * T_;

  v8f acc[2][4] = {{{}, {}, {}, {}}, {{}, {}, {}, {}}};
  float mx[2][8], ls[2][8];
#pragma unroll
  for (int mh = 0; mh < 2; ++mh)
#pragma unroll
    for (int j = 0; j < 8; ++j) { mx[mh][j] = -3.0e38f; ls[mh][j] = 0.0f; }

  for (int n0 = 0; n0 <= m0 + 31; n0 += 32) {
    // S tiles: [mh][key-subtile], each 16x16 f32
    v8f st[2][2] = {{{}, {}}, {{}, {}}};
#pragma unroll
    for (int kt = 0; kt < 2; ++kt) {
      const v16bf kb0 = load_frag16x32(kbase + (size_t)(n0 + kt * 16) * HD_, HD_);
      const v16bf kb1 = load_frag16x32(kbase + (size_t)(n0 + kt * 16) * HD_ + 32, HD_);
      st[0][kt] = wmma_bf16(qa[0][0], kb0, st[0][kt]);
      st[0][kt] = wmma_bf16(qa[0][1], kb1, st[0][kt]);
      st[1][kt] = wmma_bf16(qa[1][0], kb0, st[1][kt]);
      st[1][kt] = wmma_bf16(qa[1][1], kb1, st[1][kt]);
    }

    // causal mask + online softmax (row = j + 8*half within a 16-lane half;
    // shfl_xor 1/2/4/8 reductions stay inside the row)
#pragma unroll
    for (int mh = 0; mh < 2; ++mh) {
      const bool needMask = (n0 + 31) > (m0 + mh * 16);  // uniform
#pragma unroll
      for (int j = 0; j < 8; ++j) {
        const int row  = m0 + mh * 16 + j + 8 * half;
        const int key0 = n0 + col;
        const int key1 = key0 + 16;
        float a = st[mh][0][j];
        float c = st[mh][1][j];
        if (needMask) {
          if (key0 > row) a = -1.0e30f;
          if (key1 > row) c = -1.0e30f;
        }

        float mnew = fmaxf(a, c);
        mnew = fmaxf(mnew, __shfl_xor(mnew, 1, 32));
        mnew = fmaxf(mnew, __shfl_xor(mnew, 2, 32));
        mnew = fmaxf(mnew, __shfl_xor(mnew, 4, 32));
        mnew = fmaxf(mnew, __shfl_xor(mnew, 8, 32));
        mnew = fmaxf(mnew, mx[mh][j]);

        const float corr = __expf(mx[mh][j] - mnew);
        mx[mh][j] = mnew;
        const float p0 = __expf(a - mnew);
        const float p1 = __expf(c - mnew);
        float ssum = p0 + p1;
        ssum += __shfl_xor(ssum, 1, 32);
        ssum += __shfl_xor(ssum, 2, 32);
        ssum += __shfl_xor(ssum, 4, 32);
        ssum += __shfl_xor(ssum, 8, 32);
        ls[mh][j] = ls[mh][j] * corr + ssum;
        acc[mh][0][j] *= corr; acc[mh][1][j] *= corr;
        acc[mh][2][j] *= corr; acc[mh][3][j] *= corr;

        const int r = mh * 16 + j + 8 * half;   // P row-major (2x)16x32 in LDS
        lds[r * 32 + col]      = (__bf16)p0;
        lds[r * 32 + col + 16] = (__bf16)p1;
      }
    }
    __syncthreads();

    const v16bf pa0 = load_frag16x32(lds, 32);            // P rows 0..15
    const v16bf pa1 = load_frag16x32(lds + 16 * 32, 32);  // P rows 16..31
#pragma unroll
    for (int s = 0; s < 4; ++s) {
      const v16bf vb = load_frag16x32(vbase + (size_t)s * 16 * T_ + n0, T_);
      acc[0][s] = wmma_bf16(pa0, vb, acc[0][s]);
      acc[1][s] = wmma_bf16(pa1, vb, acc[1][s]);
    }
    __syncthreads();
  }

  // normalize and write y in [B][T][HQ*HD] layout (bf16) for the out-proj GEMM
#pragma unroll
  for (int mh = 0; mh < 2; ++mh) {
#pragma unroll
    for (int j = 0; j < 8; ++j) {
      const int t = m0 + mh * 16 + j + 8 * half;
      const float inv = 1.0f / ls[mh][j];
      __bf16* yrow = yg + ((size_t)b * T_ + t) * C_ + h * HD_;
      yrow[col]      = (__bf16)(acc[mh][0][j] * inv);
      yrow[col + 16] = (__bf16)(acc[mh][1][j] * inv);
      yrow[col + 32] = (__bf16)(acc[mh][2][j] * inv);
      yrow[col + 48] = (__bf16)(acc[mh][3][j] * inv);
    }
  }
}

// ---------------------------------------------------------------------------
// Kernel 3: out = y @ Wo^T + bo   (fp32 output), 32x64 tile per wave
// grid = (B*T/32, C/64), block = 32
// ---------------------------------------------------------------------------
__global__ void __launch_bounds__(32)
out_proj_kernel(const __bf16* __restrict__ y, const __bf16* __restrict__ Wob,
                const float* __restrict__ bo, float* __restrict__ out) {
  const int lane = threadIdx.x & 31;
  const int half = lane >> 4;
  const int col  = lane & 15;
  const int rt   = blockIdx.x;
  const int n0   = blockIdx.y * 64;

  const __bf16* ya = y + (size_t)rt * 32 * C_;
  const __bf16* wb = Wob + (size_t)n0 * C_;

  v8f acc[2][4] = {{{}, {}, {}, {}}, {{}, {}, {}, {}}};
#pragma unroll 2
  for (int kk = 0; kk < C_; kk += 32) {
    const v16bf a0 = load_frag16x32(ya + kk, C_);
    const v16bf a1 = load_frag16x32(ya + (size_t)16 * C_ + kk, C_);
#pragma unroll
    for (int s = 0; s < 4; ++s) {
      const v16bf w = load_frag16x32(wb + (size_t)s * 16 * C_ + kk, C_);
      acc[0][s] = wmma_bf16(a0, w, acc[0][s]);
      acc[1][s] = wmma_bf16(a1, w, acc[1][s]);
    }
  }

#pragma unroll
  for (int mh = 0; mh < 2; ++mh) {
#pragma unroll
    for (int s = 0; s < 4; ++s) {
      const int   n    = n0 + s * 16 + col;
      const float bias = bo[n];
#pragma unroll
      for (int j = 0; j < 8; ++j) {
        const int row = rt * 32 + mh * 16 + j + 8 * half;
        out[(size_t)row * C_ + n] = acc[mh][s][j] + bias;
      }
    }
  }
}

// ---------------------------------------------------------------------------
extern "C" void kernel_launch(void* const* d_in, const int* in_sizes, int n_in,
                              void* d_out, int out_size, void* d_ws, size_t ws_size,
                              hipStream_t stream) {
  const float* x    = (const float*)d_in[0];
  const float* rope = (const float*)d_in[1];
  const float* Wq   = (const float*)d_in[2];
  const float* bq   = (const float*)d_in[3];
  const float* Wk   = (const float*)d_in[4];
  const float* bk   = (const float*)d_in[5];
  const float* Wv   = (const float*)d_in[6];
  const float* bv   = (const float*)d_in[7];
  const float* Wo   = (const float*)d_in[8];
  const float* bo   = (const float*)d_in[9];
  float* out = (float*)d_out;

  // element counts
  const size_t nX  = (size_t)Bn * T_ * C_;          // 4,194,304
  const size_t nWq = (size_t)HQ_ * HD_ * C_;        // 1,048,576
  const size_t nWk = (size_t)HKV_ * HD_ * C_;       //   131,072
  const size_t nWo = (size_t)C_ * C_;               // 1,048,576
  const size_t nQ  = (size_t)Bn * HQ_ * T_ * HD_;   // 4,194,304
  const size_t nKV = (size_t)Bn * HKV_ * T_ * HD_;  //   524,288

  // workspace partition (bf16 elements)
  __bf16* ws_xb = (__bf16*)d_ws;
  __bf16* ws_wq = ws_xb + nX;
  __bf16* ws_wk = ws_wq + nWq;
  __bf16* ws_wv = ws_wk + nWk;
  __bf16* ws_wo = ws_wv + nWk;
  __bf16* ws_q  = ws_wo + nWo;
  __bf16* ws_k  = ws_q + nQ;
  __bf16* ws_vT = ws_k + nKV;
  __bf16* ws_y  = ws_vT + nKV;
  (void)ws_size; (void)in_sizes; (void)n_in; (void)out_size;

  // one-time f32 -> bf16 conversion of activations and weights
  auto cvt = [&](const float* s, __bf16* d, size_t n) {
    cvt_f32_bf16_kernel<<<dim3((unsigned)(n / (256 * 8))), 256, 0, stream>>>(s, d, (int)n);
  };
  cvt(x,  ws_xb, nX);
  cvt(Wq, ws_wq, nWq);
  cvt(Wk, ws_wk, nWk);
  cvt(Wv, ws_wv, nWk);
  cvt(Wo, ws_wo, nWo);

  const int rowTiles32 = (Bn * T_) / 32;  // 128

  dim3 gA(rowTiles32, 16 + 2 + 2);  // 128 x 20 (32-row x 64-wide tiles)
  qkv_rope_kernel<<<gA, 32, 0, stream>>>(ws_xb, rope, ws_wq, bq, ws_wk, bk,
                                         ws_wv, bv, ws_q, ws_k, ws_vT);

  dim3 gB(Bn * HQ_, T_ / 32);  // 32 x 64
  attn_kernel<<<gB, 32, 0, stream>>>(ws_q, ws_k, ws_vT, ws_y);

  dim3 gC(rowTiles32, C_ / 64);  // 128 x 16
  out_proj_kernel<<<gC, 32, 0, stream>>>(ws_y, ws_wo, bo, out);
}
